// ImplicitTransformer_89446988906997
// MI455X (gfx1250) — compile-verified, matrix operand order checked by reading, not run on hardware
//
#include <hip/hip_runtime.h>
#include <hip/hip_bf16.h>
#include <math.h>

typedef __bf16 bfloat;
typedef bfloat bf16x16 __attribute__((ext_vector_type(16)));
typedef bfloat bf16x8  __attribute__((ext_vector_type(8)));
typedef float  f32x8   __attribute__((ext_vector_type(8)));
typedef unsigned int u32x4 __attribute__((ext_vector_type(4)));
typedef int i32x4 __attribute__((ext_vector_type(4)));
typedef int i32x8 __attribute__((ext_vector_type(8)));

#define NHEADS 8
#define RELUD  2048
#define LND    512
#define EMBD   512
#define NCLS   10
#define BB     8
#define LL     384
#define TT     385
#define STATE  3072
#define DHEAD  64
#define KITERS 12
#define MT     (BB*TT)        /* 3080 rows */
#define KZ     (STATE+EMBD)   /* 3584 */
#define NZ     (RELUD+LND)    /* 2560 */
#define NQKV   (3*512)        /* 1536 */
#define KCHUNK 128            /* TDM-staged K chunk (16KB tile) */

// ---------------------------------------------------------------------------
// Build u (cls + masked embedding + sinusoidal pos-emb) in bf16 into the last
// 512 columns of BOTH ping-pong state buffers; zero the 3072 state columns.
// ---------------------------------------------------------------------------
__global__ __launch_bounds__(256)
void init_state_kernel(const int* __restrict__ tokens,
                       const int* __restrict__ seq_lens,
                       const float* __restrict__ emb,
                       const float* __restrict__ cls_token,
                       bfloat* __restrict__ xu0, bfloat* __restrict__ xu1)
{
    int m = blockIdx.x;
    int b = m / TT, t = m % TT;
    long base = (long)m * KZ;
    for (int s = threadIdx.x; s < KZ; s += 256) {
        float val = 0.0f;
        if (s >= STATE) {
            int d = s - STATE;
            if (t == 0) {
                val = cls_token[d];
            } else {
                int pos = t - 1;
                int tok = tokens[b * LL + pos];
                float e = (pos < seq_lens[b]) ? emb[(long)tok * EMBD + d] : 0.0f;
                float expo = (float)(d & ~1) * (1.0f / (float)EMBD);
                float ang  = (float)pos * __expf(-expo * 9.210340371976184f); // 10000^-expo
                float pe   = (d & 1) ? __cosf(ang) : __sinf(ang);
                val = e + pe;
            }
        }
        bfloat bv = (bfloat)val;
        xu0[base + s] = bv;
        xu1[base + s] = bv;
    }
}

// f32 -> bf16 weight conversion (done once; weights then live in L2)
__global__ __launch_bounds__(256)
void cvt_kernel(const float* __restrict__ src, bfloat* __restrict__ dst, int n)
{
    int i = blockIdx.x * 256 + threadIdx.x;
    if (i < n) dst[i] = (bfloat)src[i];
}

// ---------------------------------------------------------------------------
// TDM issue: 2-D tile (KCHUNK x 64 rows, bf16) of the weight matrix into LDS.
// D# built per CDNA5 ISA ch.8: group0 {count, lds_addr, global_addr, type=2},
// group1 {data_size=2B, tensor dims, tile dims, dim0 stride}, groups2/3 zero.
// Issued by one wave per workgroup; tracked by TENSORcnt.
// (6-arg builtin flavor: g0, g1, g2, g3, ext-group, cpol.)
// ---------------------------------------------------------------------------
__device__ __forceinline__
void tdm_load_btile(const bfloat* wbase, long row0, long kc, int nrows,
                    long ld, unsigned lds_addr)
{
    unsigned long long ga =
        (unsigned long long)(size_t)(wbase + row0 * ld + kc);
    u32x4 g0 = { 1u,                                     // count=1, user mode
                 lds_addr,                               // LDS byte address
                 (unsigned)(ga & 0xFFFFFFFFu),
                 (unsigned)((ga >> 32) & 0x1FFFFFFu) | (2u << 30) }; // type=2
    unsigned dim0 = (unsigned)ld;       // tensor_dim0 (elements)
    unsigned dim1 = (unsigned)nrows;    // tensor_dim1 (rows from tile start)
    i32x8 g1 = { (int)(1u << 16),                        // data_size = 2 bytes
                 (int)((dim0 & 0xFFFFu) << 16),          // [63:48] dim0 lo16
                 (int)((dim0 >> 16) | ((dim1 & 0xFFFFu) << 16)), // dim0 hi | dim1 lo
                 (int)((dim1 >> 16) | ((unsigned)KCHUNK << 16)), // dim1 hi | tile_dim0
                 64,                                     // tile_dim1 = 64 rows
                 (int)dim0,                              // tensor_dim0_stride lo32
                 0, 0 };
    i32x4 gz4 = { 0, 0, 0, 0 };
    i32x8 gz8 = { 0, 0, 0, 0, 0, 0, 0, 0 };
    __builtin_amdgcn_tensor_load_to_lds(g0, g1, gz4, gz4, gz8, 0);
}

// ---------------------------------------------------------------------------
// z = XU(3080x3584) @ A^T(3584x2560) + b  via V_WMMA_F32_16X16X32_BF16.
// 8 waves/block, block tile 128x64, wave tile 16x64 (4 accumulators).
// Weight tile TDM-staged into double-buffered LDS (16KB x2); activations
// streamed from global; A- and B-fragments preloaded into distinct registers
// so DS/VMEM waits are partial and overlap WMMA issue.
// Epilogue: cols <2048 -> relu -> bf16 next-state; cols 2048..2559 -> z_ln.
// ---------------------------------------------------------------------------
__global__ __launch_bounds__(256)
void z_gemm_kernel(const bfloat* __restrict__ xu,   // MT x KZ
                   const bfloat* __restrict__ Abf,  // NZ x KZ (row-major = B^T)
                   const float*  __restrict__ bias, // NZ
                   bfloat* __restrict__ xu_next,    // MT x KZ
                   float*  __restrict__ z_ln)       // MT x LND
{
    __shared__ bfloat btile[2][64 * KCHUNK];        // 32 KB of 320 KB WGP LDS
    int wave = threadIdx.x >> 5, lane = threadIdx.x & 31;
    int tile_m = blockIdx.x * 128 + wave * 16;
    int tile_n = blockIdx.y * 64;
    int lrow   = lane & 15;
    int khalf  = (lane >> 4) * 8;          // K-half select per ISA A/B layout

    int arow = tile_m + lrow; if (arow > MT - 1) arow = MT - 1;
    const bfloat* aptr = xu + (long)arow * KZ;
    unsigned lds0 = (unsigned)(size_t)(&btile[0][0]);   // low 32 bits = LDS offset

    const int NCH = KZ / KCHUNK;                        // 28 chunks
    if (wave == 0) {
        tdm_load_btile(Abf, tile_n, 0, NZ, KZ, lds0);
        __builtin_amdgcn_s_wait_tensorcnt(0);
    }
    __syncthreads();

    f32x8 acc[4] = {};
    for (int ch = 0; ch < NCH; ++ch) {
        // kick DMA for next chunk into the other LDS buffer (overlaps compute)
        if (wave == 0 && ch + 1 < NCH)
            tdm_load_btile(Abf, tile_n, (long)(ch + 1) * KCHUNK, NZ, KZ,
                           lds0 + (unsigned)(((ch + 1) & 1) * 64 * KCHUNK * 2));

        long kbase = (long)ch * KCHUNK;
        __builtin_prefetch(aptr + kbase + 2 * KCHUNK, 0, 1); // stream activations
        // preload all 4 A-fragments of this chunk (8 x b128 global loads)
        union frag { bf16x16 v; bf16x8 h[2]; };
        frag af[4];
#pragma unroll
        for (int s = 0; s < 4; ++s) {
            af[s].h[0] = *(const bf16x8*)(aptr + kbase + s * 32 + khalf);
            af[s].h[1] = *(const bf16x8*)(aptr + kbase + s * 32 + 16 + khalf);
        }
        const bfloat* bbuf = &btile[ch & 1][0];
#pragma unroll
        for (int s = 0; s < 4; ++s) {
            int ksub = s * 32;
            // preload the 4 B-fragments of this sub-step into distinct regs:
            // 8 ds_load_b128 issue as a clause, WMMAs then need only partial
            // dscnt waits instead of a full drain per WMMA.
            frag bfr[4];
#pragma unroll
            for (int j = 0; j < 4; ++j) {
                const bfloat* bp = bbuf + (j * 16 + lrow) * KCHUNK + ksub;
                bfr[j].h[0] = *(const bf16x8*)(bp + khalf);
                bfr[j].h[1] = *(const bf16x8*)(bp + 16 + khalf);
            }
#pragma unroll
            for (int j = 0; j < 4; ++j)
                acc[j] = __builtin_amdgcn_wmma_f32_16x16x32_bf16(
                    false, af[s].v, false, bfr[j].v, (short)0, acc[j], false, false);
        }
        if (wave == 0 && ch + 1 < NCH)
            __builtin_amdgcn_s_wait_tensorcnt(0);   // next buffer landed
        __syncthreads();                            // safe to flip buffers
    }

    int orow0 = tile_m + 8 * (lane >> 4);  // C/D layout: row = r + 8*(lane>>4)
#pragma unroll
    for (int j = 0; j < 4; ++j) {
        int col = tile_n + j * 16 + lrow;
        float bv = bias[col];
#pragma unroll
        for (int r = 0; r < 8; ++r) {
            int row = orow0 + r;
            if (row >= MT) continue;
            float z = acc[j][r] + bv;
            if (col < RELUD)
                xu_next[(long)row * KZ + col] = (bfloat)fmaxf(z, 0.0f);
            else
                z_ln[(long)row * LND + (col - RELUD)] = z;
        }
    }
}

// ---------------------------------------------------------------------------
// QKV = ln_x(3080x512) @ A_qkv^T(512x1536), f32 out. Same WMMA structure,
// weights direct from L2 (whole A_qkv is only 1.6MB); B-frags preloaded.
// ---------------------------------------------------------------------------
__global__ __launch_bounds__(256)
void qkv_gemm_kernel(const bfloat* __restrict__ xu,
                     const bfloat* __restrict__ Wq,   // NQKV x LND
                     float* __restrict__ outq)        // MT x NQKV
{
    int wave = threadIdx.x >> 5, lane = threadIdx.x & 31;
    int tile_m = blockIdx.x * 128 + wave * 16;
    int tile_n = blockIdx.y * 64;
    int lrow   = lane & 15;
    int khalf  = (lane >> 4) * 8;

    int arow = tile_m + lrow; if (arow > MT - 1) arow = MT - 1;
    const bfloat* aptr = xu + (long)arow * KZ + RELUD;

    f32x8 acc[4] = {};
    for (int kc = 0; kc < LND; kc += 32) {
        union frag { bf16x16 v; bf16x8 h[2]; };
        frag af;
        af.h[0] = *(const bf16x8*)(aptr + kc + khalf);
        af.h[1] = *(const bf16x8*)(aptr + kc + 16 + khalf);
        frag bfr[4];
#pragma unroll
        for (int j = 0; j < 4; ++j) {
            const bfloat* bptr = Wq + (long)(tile_n + j * 16 + lrow) * LND + kc;
            bfr[j].h[0] = *(const bf16x8*)(bptr + khalf);
            bfr[j].h[1] = *(const bf16x8*)(bptr + 16 + khalf);
        }
#pragma unroll
        for (int j = 0; j < 4; ++j)
            acc[j] = __builtin_amdgcn_wmma_f32_16x16x32_bf16(
                false, af.v, false, bfr[j].v, (short)0, acc[j], false, false);
    }

    int orow0 = tile_m + 8 * (lane >> 4);
#pragma unroll
    for (int j = 0; j < 4; ++j) {
        int col = tile_n + j * 16 + lrow;
#pragma unroll
        for (int r = 0; r < 8; ++r) {
            int row = orow0 + r;
            if (row < MT) outq[(long)row * NQKV + col] = acc[j][r];
        }
    }
}

// ---------------------------------------------------------------------------
// LayerNorm over the 512-wide slice (N_LN == 1), writes bf16 into next state.
// ---------------------------------------------------------------------------
__global__ __launch_bounds__(256)
void ln_kernel(const float* __restrict__ z_ln, bfloat* __restrict__ xu_next)
{
    __shared__ float ssum[256], ssq[256];
    int row = blockIdx.x, tid = threadIdx.x;
    float v0 = z_ln[(long)row * LND + tid];
    float v1 = z_ln[(long)row * LND + tid + 256];
    ssum[tid] = v0 + v1;
    ssq[tid]  = v0 * v0 + v1 * v1;
    __syncthreads();
    for (int off = 128; off > 0; off >>= 1) {
        if (tid < off) { ssum[tid] += ssum[tid + off]; ssq[tid] += ssq[tid + off]; }
        __syncthreads();
    }
    float mu   = ssum[0] * (1.0f / LND);
    float var  = ssq[0] * (1.0f / LND) - mu * mu;
    float rstd = rsqrtf(var + 1e-5f);
    long obase = (long)row * KZ + RELUD;
    xu_next[obase + tid]       = (bfloat)((v0 - mu) * rstd);
    xu_next[obase + tid + 256] = (bfloat)((v1 - mu) * rstd);
}

// ---------------------------------------------------------------------------
// Attention: one wave32 per (b,h,t) row; online (flash) softmax over keys
// s in [0, seq_len] (masked keys get exactly-zero weight, matching -1e9).
// Each lane owns head dims {lane, lane+32}; scores via xor-butterfly reduce.
// ---------------------------------------------------------------------------
__global__ __launch_bounds__(256)
void attn_kernel(const float* __restrict__ qkv,
                 const int* __restrict__ seq_lens,
                 bfloat* __restrict__ xu_next)
{
    int wave = threadIdx.x >> 5, lane = threadIdx.x & 31;
    int gid = blockIdx.x * 8 + wave;
    if (gid >= BB * NHEADS * TT) return;
    int b = gid / (NHEADS * TT);
    int rem = gid % (NHEADS * TT);
    int h = rem / TT, t = rem % TT;

    long qbase = (long)(b * TT + t) * NQKV + h * (3 * DHEAD);
    float q0 = qkv[qbase + lane];
    float q1 = qkv[qbase + lane + 32];
    int smax = seq_lens[b];                 // keys allowed: s <= seq_len

    float mrun = -INFINITY, lrun = 0.0f, a0 = 0.0f, a1 = 0.0f;
    const float scale = 0.125f;             // 1/sqrt(64)
    for (int s = 0; s <= smax; ++s) {
        long kb = (long)(b * TT + s) * NQKV + h * (3 * DHEAD) + DHEAD;
        float p = q0 * qkv[kb + lane] + q1 * qkv[kb + lane + 32];
        for (int off = 16; off > 0; off >>= 1) p += __shfl_xor(p, off, 32);
        float sc   = p * scale;
        float mnew = fmaxf(mrun, sc);
        float corr = __expf(mrun - mnew);
        float w    = __expf(sc - mnew);
        float v0   = qkv[kb + DHEAD + lane];
        float v1   = qkv[kb + DHEAD + lane + 32];
        lrun = lrun * corr + w;
        a0   = a0 * corr + w * v0;
        a1   = a1 * corr + w * v1;
        mrun = mnew;
    }
    float inv = 1.0f / lrun;
    long ob = (long)(b * TT + t) * KZ + NZ + h * DHEAD;
    xu_next[ob + lane]      = (bfloat)(a0 * inv);
    xu_next[ob + lane + 32] = (bfloat)(a1 * inv);
}

// ---------------------------------------------------------------------------
// Classifier head: out[b,c] = X[b,0,:3072] . W_cls[c] + b_cls[c]. 80 waves.
// ---------------------------------------------------------------------------
__global__ __launch_bounds__(256)
void head_kernel(const bfloat* __restrict__ xu,
                 const float* __restrict__ Wcls,
                 const float* __restrict__ bcls,
                 float* __restrict__ out)
{
    int wave = threadIdx.x >> 5, lane = threadIdx.x & 31;
    int gid = blockIdx.x * 8 + wave;
    if (gid >= BB * NCLS) return;
    int b = gid / NCLS, c = gid % NCLS;
    const bfloat* xrow = xu + (long)(b * TT) * KZ;
    const float*  wrow = Wcls + (long)c * STATE;
    float acc = 0.0f;
    for (int s = lane; s < STATE; s += 32)
        acc += (float)xrow[s] * wrow[s];
    for (int off = 16; off > 0; off >>= 1) acc += __shfl_xor(acc, off, 32);
    if (lane == 0) out[b * NCLS + c] = acc + bcls[c];
}

// ---------------------------------------------------------------------------
extern "C" void kernel_launch(void* const* d_in, const int* in_sizes, int n_in,
                              void* d_out, int out_size, void* d_ws, size_t ws_size,
                              hipStream_t stream)
{
    const int*   tokens   = (const int*)  d_in[0];
    const int*   seq_lens = (const int*)  d_in[1];
    const float* emb      = (const float*)d_in[2];
    const float* A        = (const float*)d_in[3];
    const float* bvec     = (const float*)d_in[4];
    const float* A_qkv    = (const float*)d_in[5];
    const float* cls_tok  = (const float*)d_in[6];
    const float* W_cls    = (const float*)d_in[7];
    const float* b_cls    = (const float*)d_in[8];
    float* out = (float*)d_out;
    (void)in_sizes; (void)n_in; (void)out_size; (void)ws_size;

    char* ws = (char*)d_ws;
    size_t off = 0;
    auto carve = [&](size_t bytes) -> void* {
        off = (off + 255) & ~(size_t)255;
        void* p = ws + off;
        off += bytes;
        return p;
    };
    bfloat* xu0  = (bfloat*)carve((size_t)MT * KZ * 2);        // ping
    bfloat* xu1  = (bfloat*)carve((size_t)MT * KZ * 2);        // pong
    bfloat* Abf  = (bfloat*)carve((size_t)NZ * KZ * 2);        // A in bf16
    bfloat* Aqbf = (bfloat*)carve((size_t)NQKV * LND * 2);     // A_qkv in bf16
    float*  qkvb = (float*) carve((size_t)MT * NQKV * 4);      // QKV activations
    float*  zln  = (float*) carve((size_t)MT * LND * 4);       // raw z for LN

    init_state_kernel<<<MT, 256, 0, stream>>>(tokens, seq_lens, emb, cls_tok, xu0, xu1);
    cvt_kernel<<<(NZ * KZ + 255) / 256, 256, 0, stream>>>(A, Abf, NZ * KZ);
    cvt_kernel<<<(NQKV * LND + 255) / 256, 256, 0, stream>>>(A_qkv, Aqbf, NQKV * LND);

    bfloat* cur = xu0; bfloat* nxt = xu1;
    for (int it = 0; it < KITERS; ++it) {
        qkv_gemm_kernel<<<dim3(25, 24), 256, 0, stream>>>(cur, Aqbf, qkvb);       // old ln_x
        z_gemm_kernel  <<<dim3(25, 40), 256, 0, stream>>>(cur, Abf, bvec, nxt, zln);
        ln_kernel      <<<MT, 256, 0, stream>>>(zln, nxt);
        attn_kernel    <<<MT, 256, 0, stream>>>(qkvb, seq_lens, nxt);
        bfloat* tmp = cur; cur = nxt; nxt = tmp;                                  // ping-pong
    }
    head_kernel<<<10, 256, 0, stream>>>(cur, W_cls, b_cls, out);
}
